// JitBiGRU_25623774888609
// MI455X (gfx1250) — matrix operand order repaired and checked by reading, compiled, not run on hardware
//
#include <hip/hip_runtime.h>
#include <hip/hip_bf16.h>

// Problem constants (from reference): B=64, T=512, D_IN=256, H=1024, G=3H=3072
#define B_   64
#define T_   512
#define DIN_ 256
#define H_   1024
#define G3_  3072

typedef __attribute__((ext_vector_type(16))) __bf16 v16bf;
typedef __attribute__((ext_vector_type(8)))  float  v8f;

union Frag16 { v16bf v; unsigned u[8]; };

__device__ __forceinline__ unsigned short f2bf(float f) {
    unsigned u = __builtin_bit_cast(unsigned, f);
    u += 0x7FFFu + ((u >> 16) & 1u);          // round-to-nearest-even
    return (unsigned short)(u >> 16);
}
__device__ __forceinline__ unsigned pack2(float lo, float hi) {
    return (unsigned)f2bf(lo) | ((unsigned)f2bf(hi) << 16);
}
__device__ __forceinline__ float sigmoidf_(float x) { return 1.0f / (1.0f + __expf(-x)); }
__device__ __forceinline__ float tanh_(float x) {
    float e = __expf(-2.0f * x);
    return (1.0f - e) / (1.0f + e);
}

// ------------- packed fp32 -> bf16 conversion (2 elements / thread) -------------
__global__ void cvt_f32_bf16_pk(const float* __restrict__ src,
                                unsigned* __restrict__ dst, int npairs) {
    int i = blockIdx.x * blockDim.x + threadIdx.x;
    if (i < npairs) {
        float2 f = ((const float2*)src)[i];
        dst[i] = pack2(f.x, f.y);
    }
}

// ---------------- input GEMM: gi[dir][t][b][3H] = x @ W_ih^T + b_ih ----------------
// x: [B][T][din] bf16 (pre-converted; backward dir reads t_src = T-1-t). W: [3H][din] bf16.
// Block = 256 thr = 8 waves; block tile = 32 rows x 128 cols; wave -> 32x16 (2 accum,
// shared B fragment -> 2 independent WMMAs per k-step).
__global__ void gi_gemm(const unsigned short* __restrict__ x, int din,
                        const unsigned short* __restrict__ w_f,
                        const unsigned short* __restrict__ w_b,
                        const float* __restrict__ bih_f,
                        const float* __restrict__ bih_b,
                        float* __restrict__ gi) {
    const int dir  = blockIdx.z;
    const unsigned short* w  = dir ? w_b  : w_f;
    const float*        bih  = dir ? bih_b : bih_f;
    const int lane = threadIdx.x & 31;
    const int wave = threadIdx.x >> 5;
    const int m0   = blockIdx.x * 32;
    const int n0   = blockIdx.y * 128 + wave * 16;
    const int half = lane >> 4;
    const int l15  = lane & 15;

    // A rows for this lane (tile rows all share one timestep: 32 | 64)
    const int t_out = m0 >> 6;              // / B_
    const int t_src = dir ? (T_ - 1 - t_out) : t_out;
    const int bb0   = (m0 & 63) + l15;
    const unsigned* arow0 = (const unsigned*)(x + ((size_t)bb0 * T_ + t_src) * din);
    const unsigned* arow1 = (const unsigned*)(x + ((size_t)(bb0 + 16) * T_ + t_src) * din);
    // B column for this lane
    const int ncol = n0 + l15;
    const unsigned* wrow = (const unsigned*)(w + (size_t)ncol * din);

    v8f acc0 = {}, acc1 = {};
    for (int k0 = 0; k0 < din; k0 += 32) {
        Frag16 a0, a1, b;
        const int kb = (k0 + half * 16) >> 1;
#pragma unroll
        for (int i = 0; i < 4; ++i) {           // A 16x32: lane=M, regs = contiguous K pairs (k = 16*half + s)
            a0.u[i]     = arow0[kb + i];
            a0.u[i + 4] = arow0[kb + 4 + i];
            a1.u[i]     = arow1[kb + i];
            a1.u[i + 4] = arow1[kb + 4 + i];
        }
        const int kw = (k0 + half * 16) >> 1;
#pragma unroll
        for (int i = 0; i < 8; ++i)             // B 32x16: lane=N, regs = K pairs
            b.u[i] = wrow[kw + i];
        acc0 = __builtin_amdgcn_wmma_f32_16x16x32_bf16(false, a0.v, false, b.v,
                                                       (short)0, acc0, false, false);
        acc1 = __builtin_amdgcn_wmma_f32_16x16x32_bf16(false, a1.v, false, b.v,
                                                       (short)0, acc1, false, false);
    }
    const float biasv = bih[ncol];
#pragma unroll
    for (int r = 0; r < 8; ++r) {               // C: M = half*8+r, N = l15
        int mm = m0 + half * 8 + r;
        int t = mm >> 6, b = mm & 63;
        gi[(((size_t)dir * T_ + t) * B_ + b) * (size_t)G3_ + ncol] = acc0[r] + biasv;
        mm += 16;
        t = mm >> 6; b = mm & 63;
        gi[(((size_t)dir * T_ + t) * B_ + b) * (size_t)G3_ + ncol] = acc1[r] + biasv;
    }
}

// ---------------- one recurrent step (both directions) ----------------
// grid = (H/16, 2), block = 128 (4 waves). Wave: rows b = wave*16.., cols j0..j0+15.
// Three shared-A WMMA accumulators (r,z,n gates) over K = H.
__global__ void gru_step(const unsigned short* __restrict__ whh_f,
                         const unsigned short* __restrict__ whh_b,
                         const float* __restrict__ bhh_f,
                         const float* __restrict__ bhh_b,
                         const float* __restrict__ gi,            // [2][T][B][3H]
                         float* __restrict__ h,                   // [2][B][H] fp32
                         const unsigned short* __restrict__ hb_in,// [2][B][H] bf16
                         unsigned short* __restrict__ hb_out,     // [2][B][H] bf16
                         float* __restrict__ out,                 // [B][T][H] (+=)
                         int t) {
    const int dir = blockIdx.y;
    const unsigned short* whh = dir ? whh_b : whh_f;
    const float*          bhh = dir ? bhh_b : bhh_f;
    const int lane = threadIdx.x & 31;
    const int wave = threadIdx.x >> 5;
    const int half = lane >> 4;
    const int l15  = lane & 15;
    const int j0   = blockIdx.x * 16;

    const unsigned* hrow = (const unsigned*)(hb_in + ((size_t)dir * B_ + wave * 16 + l15) * H_);
    const unsigned* wr = (const unsigned*)(whh + (size_t)(0 * H_ + j0 + l15) * H_);
    const unsigned* wz = (const unsigned*)(whh + (size_t)(1 * H_ + j0 + l15) * H_);
    const unsigned* wn = (const unsigned*)(whh + (size_t)(2 * H_ + j0 + l15) * H_);

    v8f ar = {}, az = {}, an = {};
    for (int k0 = 0; k0 < H_; k0 += 32) {
        Frag16 a, br, bz, bn;
        const int kb = (k0 + half * 16) >> 1;
#pragma unroll
        for (int i = 0; i < 4; ++i) {
            a.u[i]     = hrow[kb + i];
            a.u[i + 4] = hrow[kb + 4 + i];
        }
        const int kw = (k0 + half * 16) >> 1;
#pragma unroll
        for (int i = 0; i < 8; ++i) {
            br.u[i] = wr[kw + i];
            bz.u[i] = wz[kw + i];
            bn.u[i] = wn[kw + i];
        }
        ar = __builtin_amdgcn_wmma_f32_16x16x32_bf16(false, a.v, false, br.v, (short)0, ar, false, false);
        az = __builtin_amdgcn_wmma_f32_16x16x32_bf16(false, a.v, false, bz.v, (short)0, az, false, false);
        an = __builtin_amdgcn_wmma_f32_16x16x32_bf16(false, a.v, false, bn.v, (short)0, an, false, false);
    }
    const int jcol = j0 + l15;
    const float br_b = bhh[jcol], bz_b = bhh[H_ + jcol], bn_b = bhh[2 * H_ + jcol];
#pragma unroll
    for (int r = 0; r < 8; ++r) {
        const int b = wave * 16 + half * 8 + r;
        const float* gir = gi + (((size_t)dir * T_ + t) * B_ + b) * (size_t)G3_;
        float i_r = gir[jcol];
        float i_z = gir[H_ + jcol];
        float i_n = gir[2 * H_ + jcol];
        float rr = sigmoidf_(i_r + ar[r] + br_b);
        float zz = sigmoidf_(i_z + az[r] + bz_b);
        float nn = tanh_(i_n + rr * (an[r] + bn_b));
        size_t hoff = ((size_t)dir * B_ + b) * H_ + jcol;
        float hp   = h[hoff];
        float hnew = nn + zz * (hp - nn);
        h[hoff]      = hnew;
        hb_out[hoff] = f2bf(hnew);
        atomicAdd(out + ((size_t)b * T_ + t) * H_ + jcol, hnew);  // fwd+bwd sum
    }
}

extern "C" void kernel_launch(void* const* d_in, const int* in_sizes, int n_in,
                              void* d_out, int out_size, void* d_ws, size_t ws_size,
                              hipStream_t stream) {
    (void)in_sizes; (void)n_in; (void)out_size; (void)ws_size;
    const float* x = (const float*)d_in[0];
    const float* wih[4] = {(const float*)d_in[1], (const float*)d_in[5],
                           (const float*)d_in[9], (const float*)d_in[13]};
    const float* whh[4] = {(const float*)d_in[2], (const float*)d_in[6],
                           (const float*)d_in[10], (const float*)d_in[14]};
    const float* bih[4] = {(const float*)d_in[3], (const float*)d_in[7],
                           (const float*)d_in[11], (const float*)d_in[15]};
    const float* bhh[4] = {(const float*)d_in[4], (const float*)d_in[8],
                           (const float*)d_in[12], (const float*)d_in[16]};

    // ---- workspace carve (all offsets 256B-aligned) ----
    char* ws = (char*)d_ws;
    size_t off = 0;
    auto take = [&](size_t bytes) -> void* {
        void* p = ws + off;
        off = (off + bytes + 255) & ~(size_t)255;
        return p;
    };
    const size_t wih_n[4] = {(size_t)G3_ * DIN_, (size_t)G3_ * DIN_,
                             (size_t)G3_ * H_,   (size_t)G3_ * H_};
    unsigned short* wihbf[4];
    unsigned short* whhbf[4];
    for (int i = 0; i < 4; ++i) wihbf[i] = (unsigned short*)take(wih_n[i] * 2);
    for (int i = 0; i < 4; ++i) whhbf[i] = (unsigned short*)take((size_t)G3_ * H_ * 2);
    float* gi   = (float*)take((size_t)2 * T_ * B_ * G3_ * 4);   // ~805 MB
    float* xmid = (float*)take((size_t)B_ * T_ * H_ * 4);        // ~134 MB
    unsigned short* xbf = (unsigned short*)take((size_t)B_ * T_ * H_ * 2); // ~67 MB (reused)
    float* hbuf = (float*)take((size_t)2 * B_ * H_ * 4);
    unsigned short* hbb = (unsigned short*)take((size_t)2 * 2 * B_ * H_ * 2); // [parity][dir][B][H]
    const size_t hb_stride = (size_t)2 * B_ * H_;

    // ---- weight conversion (bf16, layout preserved) ----
    for (int i = 0; i < 4; ++i) {
        int np = (int)(wih_n[i] / 2);
        cvt_f32_bf16_pk<<<(np + 255) / 256, 256, 0, stream>>>(wih[i], (unsigned*)wihbf[i], np);
        np = G3_ * H_ / 2;
        cvt_f32_bf16_pk<<<(np + 255) / 256, 256, 0, stream>>>(whh[i], (unsigned*)whhbf[i], np);
    }

    const dim3 gGrid((T_ * B_) / 32, G3_ / 128, 2);
    const dim3 sGrid(H_ / 16, 2);

    // ---- layer 0 ----
    {
        int np = B_ * T_ * DIN_ / 2;
        cvt_f32_bf16_pk<<<(np + 255) / 256, 256, 0, stream>>>(x, (unsigned*)xbf, np);
    }
    gi_gemm<<<gGrid, 256, 0, stream>>>(xbf, DIN_, wihbf[0], wihbf[1], bih[0], bih[1], gi);
    hipMemsetAsync(xmid, 0, (size_t)B_ * T_ * H_ * 4, stream);
    hipMemsetAsync(hbuf, 0, (size_t)2 * B_ * H_ * 4, stream);
    hipMemsetAsync(hbb, 0, (size_t)2 * 2 * B_ * H_ * 2, stream);
    for (int t = 0; t < T_; ++t) {
        gru_step<<<sGrid, 128, 0, stream>>>(whhbf[0], whhbf[1], bhh[0], bhh[1], gi,
                                            hbuf, hbb + (t & 1) * hb_stride,
                                            hbb + (1 - (t & 1)) * hb_stride, xmid, t);
    }

    // ---- layer 1 ----
    {
        int np = B_ * T_ * H_ / 2;
        cvt_f32_bf16_pk<<<(np + 255) / 256, 256, 0, stream>>>(xmid, (unsigned*)xbf, np);
    }
    gi_gemm<<<gGrid, 256, 0, stream>>>(xbf, H_, wihbf[2], wihbf[3], bih[2], bih[3], gi);
    hipMemsetAsync(d_out, 0, (size_t)B_ * T_ * H_ * 4, stream);
    hipMemsetAsync(hbuf, 0, (size_t)2 * B_ * H_ * 4, stream);
    hipMemsetAsync(hbb, 0, (size_t)2 * 2 * B_ * H_ * 2, stream);
    for (int t = 0; t < T_; ++t) {
        gru_step<<<sGrid, 128, 0, stream>>>(whhbf[2], whhbf[3], bhh[2], bhh[3], gi,
                                            hbuf, hbb + (t & 1) * hb_stride,
                                            hbb + (1 - (t & 1)) * hb_stride,
                                            (float*)d_out, t);
    }
}
